// optimNet_44057774522892
// MI455X (gfx1250) — matrix operand order repaired and checked by reading, not compiled, not run on hardware
//
#include <hip/hip_runtime.h>

typedef _Float16 h16;
typedef __attribute__((ext_vector_type(16))) _Float16 v16h;
typedef __attribute__((ext_vector_type(8)))  _Float16 v8h;
typedef __attribute__((ext_vector_type(8)))  float    v8f;

// ---------------- WMMA fragment helpers (contiguous k order, same for A and B) ----------------
// A (16xK f16, row-major, leading dim lda): lane<16 -> row=lane&15, K k0..k0+15
//                                           lane>=16 -> same rows,  K k0+16..k0+31
__device__ __forceinline__ v16h load_a_frag(const h16* __restrict__ A, int lda,
                                            int m0, int k0, int lane) {
  int r = lane & 15, h = lane >> 4;
  const h16* p = A + (size_t)(m0 + r) * lda + (k0 + h * 16);
  v8h lo = *(const v8h*)p;
  v8h hi = *(const v8h*)(p + 8);
  v16h a;
#pragma unroll
  for (int i = 0; i < 8; ++i) { a[i] = lo[i]; a[i + 8] = hi[i]; }
  return a;
}

// B (Kx16 f16) supplied as Bt[N][K]: lane<16 -> col=lane, K k0..k0+15; lane>=16 -> col=lane-16, K k0+16..k0+31
__device__ __forceinline__ v16h load_b_frag(const h16* __restrict__ Bt, int ldb,
                                            int n0, int k0, int lane) {
  int r = lane & 15, h = lane >> 4;
  const h16* p = Bt + (size_t)(n0 + r) * ldb + (k0 + h * 16);
  return *(const v16h*)p;
}

// ---------------- CDNA5 async copy helpers ----------------
__device__ __forceinline__ uint32_t lds_off_u32(const void* p) {
  // generic pointers to LDS: addr[31:0] is the LDS byte offset (ISA 10.2 aperture rules)
  return (uint32_t)(uintptr_t)p;
}
__device__ __forceinline__ void async_b128(uint32_t lds_byte, const h16* g) {
  asm volatile("global_load_async_to_lds_b128 %0, %1, off" :: "v"(lds_byte), "v"(g) : "memory");
}
__device__ __forceinline__ void wait_async_le1() {
  asm volatile("s_wait_asynccnt 0x1" ::: "memory");
}

// ---------------- staged GEMM: block = 8 waves = 128 rows x 64 cols ----------------
// B-panel (64 cols x 32 K = 4KB) async-staged to LDS per K-step, double buffered.
// LDS layout: padded row stride 36 h16 (72B) to break bank conflicts.
#define SB_STRIDE 36
#define SB_BYTES  (64 * SB_STRIDE * 2)

template <int MODE>  // 0: C f32   1: Dh = (h16)relu(acc + bias[col])
__global__ __launch_bounds__(256) void k_gemm(const h16* __restrict__ A, int lda,
                                              const h16* __restrict__ Bt, int ldb,
                                              float* __restrict__ C,
                                              const float* __restrict__ bias,
                                              h16* __restrict__ Dh, int ldd,
                                              int M, int N, int K) {
  __shared__ h16 sB[2][64 * SB_STRIDE];
  const int tid  = threadIdx.x;
  const int wave = tid >> 5, lane = tid & 31;
  const int r = lane & 15, h = lane >> 4;
  const int m0  = (blockIdx.x * 8 + wave) * 16;
  const int n0s = blockIdx.y * 64;

  // staging assignment: 256 threads <-> 256 x 16B chunks of the 64x32 panel
  const int scol = tid >> 2, spart = tid & 3;
  const h16* gB = Bt + (size_t)(n0s + scol) * ldb + spart * 8;
  const uint32_t ldsW = lds_off_u32(&sB[0][0]) + (uint32_t)(scol * (SB_STRIDE * 2) + spart * 16);

  const int nK = K / 32;
  async_b128(ldsW, gB);                            // k-step 0 -> buf 0
  if (nK > 1) async_b128(ldsW + SB_BYTES, gB + 32);  // k-step 1 -> buf 1

  v8f acc0 = {}, acc1 = {}, acc2 = {}, acc3 = {};
  for (int ks = 0; ks < nK; ++ks) {
    wait_async_le1();      // own copy for buf[ks&1] retired (in-order per wave)
    __syncthreads();       // all waves' chunks of buf[ks&1] present
    const int k0 = ks * 32;
    v16h a = load_a_frag(A, lda, m0, k0, lane);
    const h16* bbuf = &sB[ks & 1][0];
    v16h b0 = *(const v16h*)(bbuf + (size_t)(0 * 16 + r) * SB_STRIDE + h * 16);
    v16h b1 = *(const v16h*)(bbuf + (size_t)(1 * 16 + r) * SB_STRIDE + h * 16);
    v16h b2 = *(const v16h*)(bbuf + (size_t)(2 * 16 + r) * SB_STRIDE + h * 16);
    v16h b3 = *(const v16h*)(bbuf + (size_t)(3 * 16 + r) * SB_STRIDE + h * 16);
    acc0 = __builtin_amdgcn_wmma_f32_16x16x32_f16(false, a, false, b0, (short)0, acc0, false, false);
    acc1 = __builtin_amdgcn_wmma_f32_16x16x32_f16(false, a, false, b1, (short)0, acc1, false, false);
    acc2 = __builtin_amdgcn_wmma_f32_16x16x32_f16(false, a, false, b2, (short)0, acc2, false, false);
    acc3 = __builtin_amdgcn_wmma_f32_16x16x32_f16(false, a, false, b3, (short)0, acc3, false, false);
    __syncthreads();       // everyone done reading buf[ks&1] before it is restaged
    if (ks + 2 < nK) async_b128(ldsW + (uint32_t)(ks & 1) * SB_BYTES, gB + (k0 + 64));
  }

  v8f* accs[4] = {&acc0, &acc1, &acc2, &acc3};
#pragma unroll
  for (int t = 0; t < 4; ++t) {
#pragma unroll
    for (int i = 0; i < 8; ++i) {
      int row = m0 + 8 * h + i;
      int col = n0s + t * 16 + r;
      if (row < M) {
        if (MODE == 0) {
          C[(size_t)row * N + col] = (*accs[t])[i];
        } else {
          float v = fmaxf((*accs[t])[i] + bias[col], 0.f);
          Dh[(size_t)row * ldd + col] = (h16)v;
        }
      }
    }
  }
}

// ef-term GEMM: A rows generated in registers: ef[e][k] = relu(t[e]*We[k]+be[k]).
// xw2[e][n] = ef[e]@W2e[:,n] + P[ei0[e]][n] + Q[ei1[e]][n]
__global__ __launch_bounds__(256) void k_ef_xw2(const float* __restrict__ t,
                                                const float* __restrict__ We,
                                                const float* __restrict__ be,
                                                const h16* __restrict__ W2et,  // [128][128]
                                                const float* __restrict__ PQ,  // [10000][256]
                                                const int* __restrict__ ei0,
                                                const int* __restrict__ ei1,
                                                float* __restrict__ xw2) {
  __shared__ float sWe[128], sbe[128];
  int tid = threadIdx.x;
  if (tid < 128) { sWe[tid] = We[tid]; sbe[tid] = be[tid]; }
  __syncthreads();
  int wave = tid >> 5, lane = tid & 31;
  int m0 = blockIdx.x * 16;  // edge tile
  int n0 = wave * 16;        // 8 waves cover N=128
  int r = lane & 15, h = lane >> 4;
  float tv = t[m0 + r];
  v8f c = {};
  for (int ks = 0; ks < 4; ++ks) {
    int k0 = ks * 32;
    v16h a;
#pragma unroll
    for (int i = 0; i < 8; ++i) {
      int ka = k0 + h * 16 + i;
      int kb = k0 + h * 16 + 8 + i;
      a[i]     = (h16)fmaxf(tv * sWe[ka] + sbe[ka], 0.f);
      a[i + 8] = (h16)fmaxf(tv * sWe[kb] + sbe[kb], 0.f);
    }
    v16h b = load_b_frag(W2et, 128, n0, k0, lane);
    c = __builtin_amdgcn_wmma_f32_16x16x32_f16(false, a, false, b, (short)0, c, false, false);
  }
#pragma unroll
  for (int i = 0; i < 8; ++i) {
    int m = m0 + 8 * h + i;
    int n = n0 + r;
    int a0 = ei0[m], a1 = ei1[m];
    xw2[(size_t)m * 128 + n] = c[i] + PQ[(size_t)a0 * 256 + n] + PQ[(size_t)a1 * 256 + 128 + n];
  }
}

// ---------------- small prep kernels ----------------
__global__ void k_f32_to_f16(const float* __restrict__ s, h16* __restrict__ d, long n) {
  long i = (long)blockIdx.x * blockDim.x + threadIdx.x;
  if (i < n) d[i] = (h16)s[i];
}
__global__ void k_w1t(const float* __restrict__ W1, h16* __restrict__ Wt) {  // [1024 n][1024 k]
  int i = blockIdx.x * blockDim.x + threadIdx.x;
  if (i >= 1024 * 1024) return;
  int n = i >> 10, k = i & 1023;
  float v = (n < 512) ? W1[(size_t)k * 512 + n] : W1[(size_t)(1024 + k) * 512 + (n - 512)];
  Wt[i] = (h16)v;
}
__global__ void k_wnt(const float* __restrict__ Wn, h16* __restrict__ Wt) {  // [128 n][512 k]
  int i = blockIdx.x * blockDim.x + threadIdx.x;
  if (i >= 128 * 512) return;
  int n = i >> 9, k = i & 511;
  Wt[i] = (h16)Wn[(size_t)k * 128 + n];
}
__global__ void k_wpqt(const float* __restrict__ W2, h16* __restrict__ Wt) {  // [256 n][640 k]
  int i = blockIdx.x * blockDim.x + threadIdx.x;
  if (i >= 256 * 640) return;
  int n = i / 640, k = i - n * 640;
  int row, col;
  if (n < 128) { row = (k < 512) ? k : 1024 + (k - 512); col = n; }
  else         { row = (k < 512) ? 512 + k : 1152 + (k - 512); col = n - 128; }
  Wt[i] = (h16)W2[(size_t)row * 128 + col];
}
__global__ void k_w2et(const float* __restrict__ W2, h16* __restrict__ Wt) {  // [128 n][128 k]
  int i = blockIdx.x * blockDim.x + threadIdx.x;
  if (i >= 128 * 128) return;
  int n = i >> 7, k = i & 127;
  Wt[i] = (h16)W2[(size_t)(1280 + k) * 128 + n];
}
__global__ void k_t(const float* __restrict__ ea, const float* __restrict__ Wet,
                    const float* __restrict__ bet, float* __restrict__ t, int E) {
  int e = blockIdx.x * blockDim.x + threadIdx.x;
  if (e < E) t[e] = ea[2 * e] * Wet[0] + ea[2 * e + 1] * Wet[1] + bet[0];
}
__global__ void k_zero_i32(int* p, int n) {
  int i = blockIdx.x * blockDim.x + threadIdx.x;
  if (i < n) p[i] = 0;
}
__global__ void k_hist(const int* __restrict__ ei1, int* cnt, int E) {
  int e = blockIdx.x * blockDim.x + threadIdx.x;
  if (e < E) atomicAdd(&cnt[ei1[e]], 1);
}
__global__ __launch_bounds__(1024) void k_scan(const int* __restrict__ cnt, int* __restrict__ offs, int n) {
  __shared__ int s[1024];
  int tid = threadIdx.x;
  int per = (n + 1023) / 1024;
  int base = tid * per;
  int sum = 0;
  for (int i = 0; i < per; ++i) { int c = base + i; if (c < n) sum += cnt[c]; }
  s[tid] = sum;
  __syncthreads();
  for (int d = 1; d < 1024; d <<= 1) {
    int v = 0;
    if (tid >= d) v = s[tid - d];
    __syncthreads();
    if (tid >= d) s[tid] += v;
    __syncthreads();
  }
  int run = (tid == 0) ? 0 : s[tid - 1];
  for (int i = 0; i < per; ++i) { int c = base + i; if (c < n) { offs[c] = run; run += cnt[c]; } }
  if (tid == 1023) offs[n] = s[1023];
}
__global__ void k_dinv(const int* __restrict__ cnt, float* __restrict__ dinv, int n) {
  int c = blockIdx.x * blockDim.x + threadIdx.x;
  if (c < n) dinv[c] = rsqrtf((float)cnt[c] + 1.f);
}
__global__ void k_fill(const int* __restrict__ ei1, const int* __restrict__ offs,
                       int* cursor, int* __restrict__ elist, int E) {
  int e = blockIdx.x * blockDim.x + threadIdx.x;
  if (e < E) {
    int c = ei1[e];
    int pos = atomicAdd(&cursor[c], 1);
    elist[offs[c] + pos] = e;
  }
}
// xw1[r] = A[ei0[r]] + B[ei1[r]] + t[r]*wlast    (AB = [A|B], 1024 cols; only rows r<10000 needed)
__global__ __launch_bounds__(256) void k_xw1(const float* __restrict__ AB, const float* __restrict__ t,
                                             const float* __restrict__ wlast,
                                             const int* __restrict__ ei0, const int* __restrict__ ei1,
                                             float* __restrict__ xw1) {
  int r = blockIdx.x;
  int i0 = ei0[r], i1 = ei1[r];
  float tv = t[r];
  for (int j = threadIdx.x; j < 512; j += blockDim.x)
    xw1[(size_t)r * 512 + j] =
        AB[(size_t)i0 * 1024 + j] + AB[(size_t)i1 * 1024 + 512 + j] + tv * wlast[j];
}
// GCN aggregation 1 + bias + relu -> f16 into X[:, :512] (ld 640)
__global__ __launch_bounds__(256) void k_agg1(const float* __restrict__ xw1,
                                              const int* __restrict__ offs, const int* __restrict__ elist,
                                              const int* __restrict__ ei0, const float* __restrict__ dinv,
                                              const float* __restrict__ b1, h16* __restrict__ X) {
  __shared__ int ss[256];
  __shared__ float sw[256];
  int c = blockIdx.x, j = threadIdx.x;
  float dc = dinv[c];
  int s0 = offs[c], s1 = offs[c + 1];
  float a0 = xw1[(size_t)c * 512 + j] * dc * dc;
  float a1 = xw1[(size_t)c * 512 + j + 256] * dc * dc;
  for (int base = s0; base < s1; base += 256) {
    int nn = s1 - base; if (nn > 256) nn = 256;
    __syncthreads();
    if (j < nn) { int e = elist[base + j]; int s = ei0[e]; ss[j] = s; sw[j] = dinv[s] * dc; }
    __syncthreads();
    for (int k = 0; k < nn; ++k) {
      const float* xr = xw1 + (size_t)ss[k] * 512;
      float w = sw[k];
      a0 += xr[j] * w;
      a1 += xr[j + 256] * w;
    }
  }
  X[(size_t)c * 640 + j]       = (h16)fmaxf(a0 + b1[j], 0.f);
  X[(size_t)c * 640 + j + 256] = (h16)fmaxf(a1 + b1[j + 256], 0.f);
}
// GCN aggregation 2 + b2 -> d_out [100000,128]
__global__ __launch_bounds__(128) void k_agg2(const float* __restrict__ xw2,
                                              const int* __restrict__ offs, const int* __restrict__ elist,
                                              const int* __restrict__ ei0, const float* __restrict__ dinv,
                                              const float* __restrict__ b2,
                                              float* __restrict__ out, int nT) {
  __shared__ int ss[128];
  __shared__ float sw[128];
  int c = blockIdx.x, j = threadIdx.x;
  float acc;
  if (c < nT) {
    float dc = dinv[c];
    int s0 = offs[c], s1 = offs[c + 1];
    acc = xw2[(size_t)c * 128 + j] * dc * dc;
    for (int base = s0; base < s1; base += 128) {
      int nn = s1 - base; if (nn > 128) nn = 128;
      __syncthreads();
      if (j < nn) { int e = elist[base + j]; int s = ei0[e]; ss[j] = s; sw[j] = dinv[s] * dc; }
      __syncthreads();
      for (int k = 0; k < nn; ++k) acc += xw2[(size_t)ss[k] * 128 + j] * sw[k];
    }
  } else {
    acc = xw2[(size_t)c * 128 + j];  // degree-1 self loop only
  }
  out[(size_t)c * 128 + j] = acc + b2[j];
}

// ---------------- host launcher ----------------
extern "C" void kernel_launch(void* const* d_in, const int* in_sizes, int n_in,
                              void* d_out, int out_size, void* d_ws, size_t ws_size,
                              hipStream_t stream) {
  const float* node_attr = (const float*)d_in[0];   // [Nn,1024]
  const float* edge_attr = (const float*)d_in[1];   // [E,2]
  const int*   eidx      = (const int*)d_in[2];     // [2,E]
  const float* Wet       = (const float*)d_in[5];
  const float* bet       = (const float*)d_in[6];
  const float* W1        = (const float*)d_in[7];   // [2049,512]
  const float* b1        = (const float*)d_in[8];
  const float* W2        = (const float*)d_in[9];   // [1408,128]
  const float* b2        = (const float*)d_in[10];
  const float* Wn        = (const float*)d_in[11];  // [512,128]
  const float* bn        = (const float*)d_in[12];
  const float* We        = (const float*)d_in[13];  // [1,128]
  const float* be        = (const float*)d_in[14];

  const int Nn = in_sizes[0] / 1024;   // 10000
  const int E  = in_sizes[1] / 2;      // 100000
  const int* ei0 = eidx;
  const int* ei1 = eidx + E;
  const float* wlast = W1 + (size_t)2048 * 512;  // last row of W1

  size_t off = 0;
  auto carve = [&](size_t bytes) -> void* {
    off = (off + 255) & ~(size_t)255;
    void* r = (char*)d_ws + off;
    off += bytes;
    return r;
  };
  h16*   nodes_h = (h16*)carve((size_t)Nn * 1024 * 2);
  h16*   W1t     = (h16*)carve((size_t)1024 * 1024 * 2);
  h16*   Wnt     = (h16*)carve((size_t)128 * 512 * 2);
  h16*   Wpqt    = (h16*)carve((size_t)256 * 640 * 2);
  h16*   W2et    = (h16*)carve((size_t)128 * 128 * 2);
  float* t       = (float*)carve((size_t)E * 4);
  int*   cnt     = (int*)carve((size_t)Nn * 4);
  int*   cursor  = (int*)carve((size_t)Nn * 4);
  int*   offs    = (int*)carve((size_t)(Nn + 1) * 4);
  int*   elist   = (int*)carve((size_t)E * 4);
  float* dinv    = (float*)carve((size_t)Nn * 4);
  float* AB      = (float*)carve((size_t)Nn * 1024 * 4);
  float* xw1     = (float*)carve((size_t)Nn * 512 * 4);
  h16*   X       = (h16*)carve((size_t)(Nn + 128) * 640 * 2);  // [out1h | nfh], padded rows for M-tail reads
  float* PQ      = (float*)carve((size_t)Nn * 256 * 4);
  float* xw2     = (float*)carve((size_t)E * 128 * 4);
  (void)ws_size; (void)n_in; (void)out_size;

  long nEl = (long)Nn * 1024;
  k_f32_to_f16<<<(nEl + 255) / 256, 256, 0, stream>>>(node_attr, nodes_h, nEl);
  k_w1t <<<(1024 * 1024 + 255) / 256, 256, 0, stream>>>(W1, W1t);
  k_wnt <<<(128 * 512 + 255) / 256, 256, 0, stream>>>(Wn, Wnt);
  k_wpqt<<<(256 * 640 + 255) / 256, 256, 0, stream>>>(W2, Wpqt);
  k_w2et<<<(128 * 128 + 255) / 256, 256, 0, stream>>>(W2, W2et);
  k_t   <<<(E + 255) / 256, 256, 0, stream>>>(edge_attr, Wet, bet, t, E);

  k_zero_i32<<<(Nn + 255) / 256, 256, 0, stream>>>(cnt, Nn);
  k_zero_i32<<<(Nn + 255) / 256, 256, 0, stream>>>(cursor, Nn);
  k_hist<<<(E + 255) / 256, 256, 0, stream>>>(ei1, cnt, E);
  k_scan<<<1, 1024, 0, stream>>>(cnt, offs, Nn);
  k_dinv<<<(Nn + 255) / 256, 256, 0, stream>>>(cnt, dinv, Nn);
  k_fill<<<(E + 255) / 256, 256, 0, stream>>>(ei1, offs, cursor, elist, E);

  const int mBlocks = (Nn / 16 + 7) / 8;  // 79 (tail waves compute padded rows, stores guarded)

  // AB = node_attr @ [W1a|W1b]   (M=Nn, N=1024, K=1024)
  k_gemm<0><<<dim3(mBlocks, 1024 / 64), 256, 0, stream>>>(nodes_h, 1024, W1t, 1024,
                                                          AB, nullptr, nullptr, 0, Nn, 1024, 1024);
  k_xw1<<<Nn, 256, 0, stream>>>(AB, t, wlast, ei0, ei1, xw1);
  k_agg1<<<Nn, 256, 0, stream>>>(xw1, offs, elist, ei0, dinv, b1, X);
  // nf = relu(out1@Wn + bn) -> f16 into X[:, 512:640]
  k_gemm<1><<<dim3(mBlocks, 128 / 64), 256, 0, stream>>>(X, 640, Wnt, 512,
                                                         nullptr, bn, X + 512, 640, Nn, 128, 512);
  // PQ = [out1|nf] @ [W2a;W2c | W2b;W2d]   (M=Nn, N=256, K=640)
  k_gemm<0><<<dim3(mBlocks, 256 / 64), 256, 0, stream>>>(X, 640, Wpqt, 640,
                                                         PQ, nullptr, nullptr, 0, Nn, 256, 640);
  // xw2[e] = ef(t[e])@W2e + P[ei0[e]] + Q[ei1[e]]
  k_ef_xw2<<<E / 16, 256, 0, stream>>>(t, We, be, W2et, PQ, ei0, ei1, xw2);
  // final aggregation + b2 -> d_out [E,128]
  k_agg2<<<E, 128, 0, stream>>>(xw2, offs, elist, ei0, dinv, b2, (float*)d_out, Nn);
}
